// MSAPairWeightedAveraging_91207925498262
// MI455X (gfx1250) — matrix operand order repaired and checked by reading, not compiled, not run on hardware
//
#include <hip/hip_runtime.h>
#include <hip/hip_bf16.h>

#define S_DIM 512
#define N_DIM 512
#define CMSA 64
#define CPAIR 128
#define CH 32
#define NH 8
#define XDIM (S_DIM * CH)      /* 16384 */
#define R_TOT (S_DIM * N_DIM)  /* 262144 */

typedef __attribute__((ext_vector_type(16))) __bf16          v16bf;
typedef __attribute__((ext_vector_type(8)))  float           v8f;
typedef __attribute__((ext_vector_type(4)))  float           f32x4;
typedef __attribute__((ext_vector_type(4)))  unsigned short  u16x4;
typedef __attribute__((ext_vector_type(8)))  unsigned short  u16x8;
typedef __attribute__((ext_vector_type(16))) unsigned short  u16x16;
typedef __attribute__((ext_vector_type(4)))  unsigned int    u32x4;
typedef __attribute__((ext_vector_type(4)))  int             i32x4;
typedef __attribute__((ext_vector_type(8)))  int             i32x8;

#if __has_builtin(__builtin_amdgcn_tensor_load_to_lds)
#define USE_TDM 1
#else
#define USE_TDM 0
#endif

static __device__ __forceinline__ unsigned short f2bf(float f) {
  unsigned u = __float_as_uint(f);
  u += 0x7FFFu + ((u >> 16) & 1u);   // round-to-nearest-even
  return (unsigned short)(u >> 16);
}
static __device__ __forceinline__ float bf2f(unsigned short h) {
  return __uint_as_float(((unsigned)h) << 16);
}
static __device__ __forceinline__ v16bf ld_frag32(const unsigned short* p) {
  u16x16 t = *(const u16x16*)p;
  return __builtin_bit_cast(v16bf, t);
}
static __device__ __forceinline__ v16bf ld_frag_lds2(const unsigned short* p0,
                                                     const unsigned short* p1) {
  u16x8 lo = *(const u16x8*)p0;
  u16x8 hi = *(const u16x8*)p1;
  u16x16 t = __builtin_shufflevector(lo, hi, 0,1,2,3,4,5,6,7,8,9,10,11,12,13,14,15);
  return __builtin_bit_cast(v16bf, t);
}

#if USE_TDM
// TDM: DMA a 2D bf16 tile (tile_d0 elems/row x tile_d1 rows, row stride
// stride0 elems) from global into LDS; pad each 16-DWORD row chunk with
// 4 DWORDs -> LDS row pitch 40 halfs, matching the WMMA staging layout.
// clang-23 toolchain: 6-arg builtin (g0, g1, g2, g3, g4, cpol).
static __device__ __forceinline__ void tdm_load_tile(unsigned lds_addr,
                                                     const void* gptr,
                                                     unsigned tile_d0, unsigned tile_d1,
                                                     unsigned tensor_d0, unsigned tensor_d1,
                                                     unsigned stride0) {
  unsigned long long ga = (unsigned long long)gptr;
  u32x4 g0;
  g0[0] = 1u;                                   // count=1, user mode
  g0[1] = lds_addr;                             // LDS byte address
  g0[2] = (unsigned)ga;                         // global addr lo
  g0[3] = (unsigned)(ga >> 32) | (2u << 30);    // global addr hi | type=2
  i32x8 g1;
  g1[0] = (1 << 16) |                           // data_size = 2 bytes
          (1 << 20) |                           // pad_enable
          (3 << 22) |                           // pad_interval: 16 DWORDs
          (3 << 25);                            // pad_amount: 4 DWORDs
  g1[1] = (int)((tensor_d0 & 0xFFFFu) << 16);
  g1[2] = (int)((tensor_d0 >> 16) | ((tensor_d1 & 0xFFFFu) << 16));
  g1[3] = (int)((tensor_d1 >> 16) | (tile_d0 << 16));
  g1[4] = (int)(tile_d1 & 0xFFFFu);
  g1[5] = (int)stride0;
  g1[6] = 0;
  g1[7] = 0;
  i32x4 z4 = {0, 0, 0, 0};
  i32x8 z8 = {0, 0, 0, 0, 0, 0, 0, 0};
  __builtin_amdgcn_tensor_load_to_lds(g0, g1, z4, z4, z8, 0);
}
#endif

// ---------------------------------------------------------------------------
// k0: pre-swizzle Wv/Wg/Wo into bf16 WMMA B-fragment layout.
// ---------------------------------------------------------------------------
__global__ __launch_bounds__(256) void k0_prep(const float* __restrict__ Wv,
                                               const float* __restrict__ Wg,
                                               const float* __restrict__ Wo,
                                               unsigned short* __restrict__ WvF,
                                               unsigned short* __restrict__ WgF,
                                               unsigned short* __restrict__ WoF) {
  int idx  = blockIdx.x * 256 + threadIdx.x;
  int arr  = idx >> 14;
  int e    = idx & 16383;
  int lane = (e >> 4) & 31;
  int hh   = e & 15;
  if (arr < 2) {                 // Wv / Wg: [64][256]
    int tile = e >> 10;
    int ks   = (e >> 9) & 1;
    int k    = ks * 32 + ((lane >> 4) << 4) + hh;
    int col  = tile * 16 + (lane & 15);
    const float* W = (arr == 0) ? Wv : Wg;
    unsigned short* D = (arr == 0) ? WvF : WgF;
    D[e] = f2bf(W[k * 256 + col]);
  } else {                       // Wo: [256][64]
    int tile = e >> 12;
    int ks   = (e >> 9) & 7;
    int k    = ks * 32 + ((lane >> 4) << 4) + hh;
    int col  = tile * 16 + (lane & 15);
    WoF[e] = f2bf(Wo[k * 64 + col]);
  }
}

// ---------------------------------------------------------------------------
// k1: msa LayerNorm + v = mn@Wv  -> vT[h][x=s*32+c][j]   (K-major for k3 B)
//                   + g = sigmoid(mn@Wg) -> g[h][i][x]
// ---------------------------------------------------------------------------
__global__ __launch_bounds__(256) void k1_ln_vg(const float* __restrict__ msa,
                                                const float* __restrict__ lnw,
                                                const float* __restrict__ lnb,
                                                const unsigned short* __restrict__ WvF,
                                                const unsigned short* __restrict__ WgF,
                                                unsigned short* __restrict__ vT,
                                                unsigned short* __restrict__ gOut) {
  __shared__ __align__(16) unsigned short At[16][72];
  __shared__ __align__(32) unsigned short Tr[4][16][16];  // per-wave transpose bounce
  int tid = threadIdx.x;
  int r0  = blockIdx.x * 16;
  int s   = r0 >> 9;
  int n0  = r0 & 511;

  int row = tid >> 4;
  int t   = tid & 15;
  f32x4 x = __builtin_nontemporal_load(
      (const f32x4*)(msa + (size_t)(r0 + row) * CMSA + t * 4));
  float sum = x.x + x.y + x.z + x.w;
  float ssq = x.x * x.x + x.y * x.y + x.z * x.z + x.w * x.w;
#pragma unroll
  for (int m = 1; m < 16; m <<= 1) {
    sum += __shfl_xor(sum, m, 16);
    ssq += __shfl_xor(ssq, m, 16);
  }
  float mu = sum * (1.0f / 64.0f);
  float rs = rsqrtf(ssq * (1.0f / 64.0f) - mu * mu + 1e-5f);
  f32x4 w4 = *(const f32x4*)(lnw + t * 4);
  f32x4 b4 = *(const f32x4*)(lnb + t * 4);
  u16x4 hv;
  hv[0] = f2bf((x.x - mu) * rs * w4.x + b4.x);
  hv[1] = f2bf((x.y - mu) * rs * w4.y + b4.y);
  hv[2] = f2bf((x.z - mu) * rs * w4.z + b4.z);
  hv[3] = f2bf((x.w - mu) * rs * w4.w + b4.w);
  *(u16x4*)&At[row][t * 4] = hv;
  __syncthreads();

  int lane = tid & 31;
  int wave = tid >> 5;
  v16bf afr[2];
#pragma unroll
  for (int ks = 0; ks < 2; ++ks) {
    const unsigned short* ap = &At[lane & 15][ks * 32 + ((lane >> 4) << 4)];
    afr[ks] = ld_frag_lds2(ap, ap + 8);
  }
#pragma unroll
  for (int q = 0; q < 4; ++q) {
    int ct = wave * 4 + q;                    // 0..15 Wv, 16..31 Wg
    const unsigned short* WF =
        (ct < 16) ? (WvF + (size_t)ct * 1024) : (WgF + (size_t)(ct - 16) * 1024);
    v8f acc = {0.f, 0.f, 0.f, 0.f, 0.f, 0.f, 0.f, 0.f};
#pragma unroll
    for (int ks = 0; ks < 2; ++ks) {
      v16bf bfr = ld_frag32(WF + ((size_t)ks * 32 + lane) * 16);
      acc = __builtin_amdgcn_wmma_f32_16x16x32_bf16(false, afr[ks], false, bfr,
                                                    (short)0, acc, false, false);
    }
    int nloc = lane & 15;
    int mofs = (lane >> 4) << 3;
    if (ct < 16) {
      int hh = ct >> 1, c0 = (ct & 1) * 16;
#pragma unroll
      for (int pp = 0; pp < 8; ++pp)
        Tr[wave][nloc][pp + mofs] = f2bf(acc[pp]);
      __builtin_amdgcn_wave_barrier();
      if (lane < 16) {
        int xx = s * 32 + c0 + lane;
        u16x16 rowv = *(const u16x16*)&Tr[wave][lane][0];
        *(u16x16*)&vT[((size_t)(hh * 16384 + xx)) * 512 + n0] = rowv;
      }
      __builtin_amdgcn_wave_barrier();
    } else {
      int ct2 = ct - 16;
      int hh = ct2 >> 1, c0 = (ct2 & 1) * 16;
      int xx = s * 32 + c0 + nloc;
#pragma unroll
      for (int pp = 0; pp < 8; ++pp) {
        int ii = n0 + pp + mofs;
        float gv = __builtin_amdgcn_rcpf(1.0f + __expf(-acc[pp]));
        gOut[((size_t)(hh * 512 + ii)) * XDIM + xx] = f2bf(gv);
      }
    }
  }
}

// ---------------------------------------------------------------------------
// k2: pair LayerNorm + b = pn@Wb + softmax over j -> w bf16 [h][i][j]
// ---------------------------------------------------------------------------
__global__ __launch_bounds__(256) void k2_bias_softmax(const float* __restrict__ pair,
                                                       const float* __restrict__ lnw,
                                                       const float* __restrict__ lnb,
                                                       const float* __restrict__ Wb,
                                                       unsigned short* __restrict__ wOut) {
  __shared__ float bbuf[512][NH];
  __shared__ float sWb[CPAIR][NH];
  int tid = threadIdx.x;
  int i   = blockIdx.x;
  for (int e = tid; e < CPAIR * NH; e += 256) sWb[e / NH][e % NH] = Wb[e];
  int l = tid & 3;
  float lw[32], lb[32];
#pragma unroll
  for (int c = 0; c < 32; ++c) { lw[c] = lnw[l * 32 + c]; lb[c] = lnb[l * 32 + c]; }
  __syncthreads();

  int quad = tid >> 2;
  for (int pass = 0; pass < 8; ++pass) {
    int j = pass * 64 + quad;
    const float* src = pair + ((size_t)i * 512 + j) * CPAIR + l * 32;
    float vals[32];
    float sum = 0.f, ssq = 0.f;
#pragma unroll
    for (int c = 0; c < 32; c += 4) {
      f32x4 v4 = __builtin_nontemporal_load((const f32x4*)(src + c));
      vals[c] = v4.x; vals[c + 1] = v4.y; vals[c + 2] = v4.z; vals[c + 3] = v4.w;
      sum += v4.x + v4.y + v4.z + v4.w;
      ssq += v4.x * v4.x + v4.y * v4.y + v4.z * v4.z + v4.w * v4.w;
    }
    sum += __shfl_xor(sum, 1, 4); sum += __shfl_xor(sum, 2, 4);
    ssq += __shfl_xor(ssq, 1, 4); ssq += __shfl_xor(ssq, 2, 4);
    float mu = sum * (1.f / 128.f);
    float rs = rsqrtf(ssq * (1.f / 128.f) - mu * mu + 1e-5f);
    float dot[NH];
#pragma unroll
    for (int h = 0; h < NH; ++h) dot[h] = 0.f;
#pragma unroll
    for (int c = 0; c < 32; ++c) {
      float ln = (vals[c] - mu) * rs * lw[c] + lb[c];
#pragma unroll
      for (int h = 0; h < NH; ++h) dot[h] += ln * sWb[l * 32 + c][h];
    }
#pragma unroll
    for (int h = 0; h < NH; ++h) {
      dot[h] += __shfl_xor(dot[h], 1, 4);
      dot[h] += __shfl_xor(dot[h], 2, 4);
    }
    if (l == 0) {
#pragma unroll
      for (int h = 0; h < NH; ++h) bbuf[j][h] = dot[h];
    }
  }
  __syncthreads();

  int wave = tid >> 5, lane = tid & 31;
  float ev[16];
  float mx = -3.4e38f;
#pragma unroll
  for (int t2 = 0; t2 < 16; ++t2) {
    float b = bbuf[t2 * 32 + lane][wave];
    ev[t2] = b;
    mx = fmaxf(mx, b);
  }
#pragma unroll
  for (int m = 1; m < 32; m <<= 1) mx = fmaxf(mx, __shfl_xor(mx, m, 32));
  float se = 0.f;
#pragma unroll
  for (int t2 = 0; t2 < 16; ++t2) { ev[t2] = __expf(ev[t2] - mx); se += ev[t2]; }
#pragma unroll
  for (int m = 1; m < 32; m <<= 1) se += __shfl_xor(se, m, 32);
  float inv = __builtin_amdgcn_rcpf(se);
#pragma unroll
  for (int t2 = 0; t2 < 16; ++t2)
    wOut[((size_t)wave * 512 + i) * 512 + t2 * 32 + lane] = f2bf(ev[t2] * inv);
}

// ---------------------------------------------------------------------------
// k3: per-head O[i,x] = sum_j W[i,j] * V[j,x]; A tile via TDM (double-buffered
// LDS), B fragments straight from L2-resident vT[h][x][j]; fused gating.
// ---------------------------------------------------------------------------
__global__ __launch_bounds__(256) void k3_attn(const unsigned short* __restrict__ wAtt,
                                               const unsigned short* __restrict__ vT,
                                               const unsigned short* __restrict__ gIn,
                                               unsigned short* __restrict__ ogOut) {
  __shared__ __align__(16) unsigned short Abuf[2][64][40];
  int tid = threadIdx.x;
  int hh  = blockIdx.z;
  int i0  = blockIdx.y * 64;
  int x0  = blockIdx.x * 128;
  int lane = tid & 31, wave = tid >> 5;
  int wr = wave >> 2, wc = wave & 3;
  v8f acc[2][2];
  v8f zro = {0.f, 0.f, 0.f, 0.f, 0.f, 0.f, 0.f, 0.f};
#pragma unroll
  for (int mt = 0; mt < 2; ++mt)
#pragma unroll
    for (int nt = 0; nt < 2; ++nt) acc[mt][nt] = zro;

  const unsigned short* aTile = wAtt + (size_t)(hh * 512 + i0) * 512;
  const unsigned short* bLane =
      vT + ((size_t)(hh * 16384 + x0 + wc * 32 + (lane & 15))) * 512 + ((lane >> 4) << 4);

#if USE_TDM
  unsigned ldsA[2];
  ldsA[0] = (unsigned)(unsigned long long)(void*)&Abuf[0][0][0];
  ldsA[1] = (unsigned)(unsigned long long)(void*)&Abuf[1][0][0];
  if (wave == 0) {
    tdm_load_tile(ldsA[0], aTile, 32u, 64u, 512u, 4096u, 512u);
    __builtin_amdgcn_s_wait_tensorcnt(0);
  }
  __syncthreads();
#else
  int ar = tid >> 2, ac = (tid & 3) * 8;
  {
    u16x8 av = *(const u16x8*)(aTile + (size_t)ar * 512 + ac);
    *(u16x8*)&Abuf[0][ar][ac] = av;
  }
  __syncthreads();
#endif

  for (int kk = 0; kk < 16; ++kk) {
    int buf = kk & 1;
    if (kk < 15) {
#if USE_TDM
      if (wave == 0)
        tdm_load_tile(ldsA[buf ^ 1], aTile + (kk + 1) * 32, 32u, 64u, 512u, 4096u, 512u);
#else
      u16x8 av = *(const u16x8*)(aTile + (size_t)ar * 512 + (kk + 1) * 32 + ac);
      *(u16x8*)&Abuf[buf ^ 1][ar][ac] = av;
#endif
    }
    v16bf af[2], bfr[2];
#pragma unroll
    for (int mt = 0; mt < 2; ++mt) {
      const unsigned short* ap =
          &Abuf[buf][wr * 32 + mt * 16 + (lane & 15)][(lane >> 4) << 4];
      af[mt] = ld_frag_lds2(ap, ap + 8);
    }
#pragma unroll
    for (int nt = 0; nt < 2; ++nt)
      bfr[nt] = ld_frag32(bLane + (size_t)nt * 16 * 512 + kk * 32);
#pragma unroll
    for (int mt = 0; mt < 2; ++mt)
#pragma unroll
      for (int nt = 0; nt < 2; ++nt)
        acc[mt][nt] = __builtin_amdgcn_wmma_f32_16x16x32_bf16(
            false, af[mt], false, bfr[nt], (short)0, acc[mt][nt], false, false);
    if (kk < 15) {
#if USE_TDM
      if (wave == 0) __builtin_amdgcn_s_wait_tensorcnt(0);
#endif
      __syncthreads();
    }
  }

  int nloc = lane & 15, mofs = (lane >> 4) << 3;
#pragma unroll
  for (int mt = 0; mt < 2; ++mt) {
#pragma unroll
    for (int nt = 0; nt < 2; ++nt) {
      int xx = x0 + wc * 32 + nt * 16 + nloc;
      int s  = xx >> 5, c = xx & 31;
#pragma unroll
      for (int pp = 0; pp < 8; ++pp) {
        int ii = i0 + wr * 32 + mt * 16 + pp + mofs;
        float gv = bf2f(__builtin_nontemporal_load(
            gIn + ((size_t)(hh * 512 + ii)) * XDIM + xx));
        ogOut[((size_t)(s * 512 + ii)) * 256 + hh * 32 + c] = f2bf(acc[mt][nt][pp] * gv);
      }
    }
  }
}

// ---------------------------------------------------------------------------
// k4: out[r,64] = og[r,256] @ Wo
// ---------------------------------------------------------------------------
__global__ __launch_bounds__(256) void k4_outproj(const unsigned short* __restrict__ og,
                                                  const unsigned short* __restrict__ WoF,
                                                  float* __restrict__ out) {
  __shared__ __align__(16) unsigned short At[128][40];
  int tid = threadIdx.x;
  int r0  = blockIdx.x * 128;
  int lane = tid & 31, wave = tid >> 5;
  int wr = wave >> 1, wc = wave & 1;
  v8f acc[2][2];
  v8f zro = {0.f, 0.f, 0.f, 0.f, 0.f, 0.f, 0.f, 0.f};
#pragma unroll
  for (int mt = 0; mt < 2; ++mt)
#pragma unroll
    for (int nt = 0; nt < 2; ++nt) acc[mt][nt] = zro;

  int ar = tid >> 1, ac = (tid & 1) * 16;
  const unsigned short* aBase = og + (size_t)(r0 + ar) * 256 + ac;

  for (int kk = 0; kk < 8; ++kk) {
    u16x16 av = *(const u16x16*)(aBase + kk * 32);
    u16x8 avlo = __builtin_shufflevector(av, av, 0, 1, 2, 3, 4, 5, 6, 7);
    u16x8 avhi = __builtin_shufflevector(av, av, 8, 9, 10, 11, 12, 13, 14, 15);
    __syncthreads();
    *(u16x8*)&At[ar][ac]     = avlo;
    *(u16x8*)&At[ar][ac + 8] = avhi;
    __syncthreads();
    v16bf af[2], bfr[2];
#pragma unroll
    for (int mt = 0; mt < 2; ++mt) {
      const unsigned short* ap = &At[wr * 32 + mt * 16 + (lane & 15)][(lane >> 4) << 4];
      af[mt] = ld_frag_lds2(ap, ap + 8);
    }
#pragma unroll
    for (int nt = 0; nt < 2; ++nt) {
      int ct = wc * 2 + nt;
      bfr[nt] = ld_frag32(WoF + ((size_t)(ct * 8 + kk) * 32 + lane) * 16);
    }
#pragma unroll
    for (int mt = 0; mt < 2; ++mt)
#pragma unroll
      for (int nt = 0; nt < 2; ++nt)
        acc[mt][nt] = __builtin_amdgcn_wmma_f32_16x16x32_bf16(
            false, af[mt], false, bfr[nt], (short)0, acc[mt][nt], false, false);
  }

  int nloc = lane & 15, mofs = (lane >> 4) << 3;
#pragma unroll
  for (int mt = 0; mt < 2; ++mt)
#pragma unroll
    for (int nt = 0; nt < 2; ++nt)
#pragma unroll
      for (int pp = 0; pp < 8; ++pp) {
        int r = r0 + wr * 32 + mt * 16 + pp + mofs;
        __builtin_nontemporal_store(acc[mt][nt][pp],
                                    out + (size_t)r * 64 + wc * 32 + nt * 16 + nloc);
      }
}

// ---------------------------------------------------------------------------
extern "C" void kernel_launch(void* const* d_in, const int* in_sizes, int n_in,
                              void* d_out, int out_size, void* d_ws, size_t ws_size,
                              hipStream_t stream) {
  (void)in_sizes; (void)n_in; (void)out_size; (void)ws_size;
  const float* msa      = (const float*)d_in[0];
  const float* pair     = (const float*)d_in[1];
  const float* ln_msa_w = (const float*)d_in[2];
  const float* ln_msa_b = (const float*)d_in[3];
  const float* ln_pr_w  = (const float*)d_in[4];
  const float* ln_pr_b  = (const float*)d_in[5];
  const float* Wv       = (const float*)d_in[6];
  const float* Wb       = (const float*)d_in[7];
  const float* Wg       = (const float*)d_in[8];
  const float* Wo       = (const float*)d_in[9];

  char* ws = (char*)d_ws;
  unsigned short* WvF = (unsigned short*)(ws);
  unsigned short* WgF = (unsigned short*)(ws + 32768);
  unsigned short* WoF = (unsigned short*)(ws + 65536);
  unsigned short* vT  = (unsigned short*)(ws + 98304);
  unsigned short* gB  = (unsigned short*)(ws + 98304 + 134217728ull);
  unsigned short* wA  = (unsigned short*)(ws + 98304 + 2ull * 134217728ull);
  unsigned short* ogB = (unsigned short*)(ws + 98304 + 2ull * 134217728ull + 4194304ull);
  float* out = (float*)d_out;

  hipLaunchKernelGGL(k0_prep, dim3(192), dim3(256), 0, stream, Wv, Wg, Wo, WvF, WgF, WoF);
  hipLaunchKernelGGL(k1_ln_vg, dim3(R_TOT / 16), dim3(256), 0, stream,
                     msa, ln_msa_w, ln_msa_b, WvF, WgF, vT, gB);
  hipLaunchKernelGGL(k2_bias_softmax, dim3(512), dim3(256), 0, stream,
                     pair, ln_pr_w, ln_pr_b, Wb, wA);
  hipLaunchKernelGGL(k3_attn, dim3(XDIM / 128, 8, 8), dim3(256), 0, stream,
                     wA, vT, gB, ogB);
  hipLaunchKernelGGL(k4_outproj, dim3(R_TOT / 128), dim3(256), 0, stream,
                     ogB, WoF, out);
}
